// Classification_2448131358812
// MI455X (gfx1250) — matrix operand order repaired and checked by reading, not compiled, not run on hardware
//
#include <hip/hip_runtime.h>
#include <hip/hip_bf16.h>
#include <stdint.h>

// ---------------- problem constants (match reference) ----------------
#define B_GRAPHS 16
#define NPG      2048
#define DEG      16
#define N_TOT    (B_GRAPHS * NPG)      // 32768
#define E_TOT    (N_TOT * DEG)         // 524288
#define NEG_SLOPE 0.2f

// ---------------- CDNA5 types ----------------
typedef __attribute__((ext_vector_type(16))) __bf16       v16bf;
typedef __attribute__((ext_vector_type(8)))  float        v8f;
typedef __attribute__((ext_vector_type(2)))  float        v2f;
typedef __attribute__((ext_vector_type(4)))  unsigned int v4u;
typedef __attribute__((ext_vector_type(8)))  int          v8i;
typedef __attribute__((ext_vector_type(4)))  int          v4i;

union FragBF { v16bf v; unsigned int u[8]; };

__device__ __forceinline__ unsigned int f32_to_bf16_bits(float f) {
  unsigned int u = __float_as_uint(f);
  u += 0x7FFFu + ((u >> 16) & 1u);   // round-to-nearest-even
  return u >> 16;
}

__device__ __forceinline__ float atomicMaxFloat(float* addr, float val) {
  float old = *addr;
  while (old < val) {
    int assumed = __float_as_int(old);
    int prev = atomicCAS((int*)addr, assumed, __float_as_int(val));
    if (prev == assumed) break;
    old = __int_as_float(prev);
  }
  return old;
}

// =====================================================================
// Tensor Data Mover: DMA a 2D f32 tile (tile_d1 rows x tile_d0 cols) from
// global into LDS.  D# layout per CDNA5 ISA ch.8 (2D: groups 2/3 zero).
// OOB rows/cols (beyond tensor_d0/d1 measured from the tile start) read 0.
// =====================================================================
__device__ __forceinline__ void tdm_load_2d_f32(
    unsigned lds_byte_off, const float* gsrc,
    unsigned tensor_d0, unsigned tensor_d1,   // remaining tensor extent (elems)
    unsigned tile_d0, unsigned tile_d1,       // tile shape (elems)
    unsigned long long stride_d0)             // row stride (elems)
{
  unsigned long long ga = (unsigned long long)(uintptr_t)gsrc;
  v4u g0;
  g0[0] = 1u;                                               // count=1 (valid)
  g0[1] = lds_byte_off;                                     // lds_addr
  g0[2] = (unsigned)(ga & 0xFFFFFFFFu);                     // global_addr lo
  g0[3] = (unsigned)((ga >> 32) & 0x01FFFFFFu) | (2u << 30);// ga hi | type=2
  v8i g1;
  g1[0] = (int)(2u << 16);                                  // data_size=4B
  g1[1] = (int)((tensor_d0 & 0xFFFFu) << 16);               // tensor_dim0 lo16
  g1[2] = (int)((tensor_d0 >> 16) | ((tensor_d1 & 0xFFFFu) << 16));
  g1[3] = (int)((tensor_d1 >> 16) | (tile_d0 << 16));       // | tile_dim0
  g1[4] = (int)(tile_d1 & 0xFFFFu);                         // tile_dim1 (dim2=0)
  g1[5] = (int)(stride_d0 & 0xFFFFFFFFu);                   // dim0_stride lo
  g1[6] = (int)((stride_d0 >> 32) & 0xFFFFu);               // dim0_stride hi
  g1[7] = 0;
  v4i gz = {0, 0, 0, 0};
#if __has_include(<hip/amd_detail/amd_gfx1250_TDM.h>)
  v8i gz8 = {0, 0, 0, 0, 0, 0, 0, 0};
  __builtin_amdgcn_tensor_load_to_lds(g0, g1, gz, gz, gz8, 0);
#else
  __builtin_amdgcn_tensor_load_to_lds(g0, g1, gz, gz, 0);
#endif
}

// =====================================================================
// Tiled GEMM:  C[M,N] = A[M,K] @ W[N,K]^T (+bias, optional relu)
// TDM stages raw f32 tiles into LDS; fragments convert f32->bf16 in regs;
// f32 accumulate via v_wmma_f32_16x16x32_bf16.
// Block = 256 threads = 8 wave32 waves; block tile 64x64; wave tile 16x32.
// Requires K % 32 == 0 (all K here are 64/128/256).
// =====================================================================
#define TM 64
#define TN 64
#define TK 32

__global__ __launch_bounds__(256) void gemm_bf16_wmma(
    const float* __restrict__ A, const float* __restrict__ W,
    const float* __restrict__ bias, float* __restrict__ C,
    int M, int N, int K, int relu)
{
  __shared__ float Asf[TM][TK];   // 8 KB: A tile rows (m-major)
  __shared__ float Bsf[TN][TK];   // 8 KB: W tile rows (n-major) = B^T

  int tid  = threadIdx.x;
  int lane = tid & 31;
  int wid  = tid >> 5;
  int m0 = blockIdx.x * TM;
  int n0 = blockIdx.y * TN;
  int waveM = (wid & 3) * 16;   // 0..48
  int waveN = (wid >> 2) * 32;  // 0 or 32

  unsigned ldsA = (unsigned)(uintptr_t)&Asf[0][0];
  unsigned ldsB = (unsigned)(uintptr_t)&Bsf[0][0];

  v8f acc0 = {};
  v8f acc1 = {};

  int mrow = waveM + (lane & 15);
  int half = lane >> 4;
  int nc0  = waveN + (lane & 15);
  int nc1  = nc0 + 16;

  for (int kk = 0; kk < K; kk += TK) {
    if (wid == 0) {
      // Tensor Data Mover: global -> LDS, no VGPR traffic, OOB rows -> 0
      tdm_load_2d_f32(ldsA, A + (size_t)m0 * K + kk,
                      (unsigned)(K - kk), (unsigned)(M - m0), TK, TM,
                      (unsigned long long)K);
      tdm_load_2d_f32(ldsB, W + (size_t)n0 * K + kk,
                      (unsigned)(K - kk), (unsigned)(N - n0), TK, TN,
                      (unsigned long long)K);
      __builtin_amdgcn_s_wait_tensorcnt(0);
    }
    __syncthreads();

    // A fragment: 16x32 (MxK) per ISA VGPR layout (K-pairs packed per VGPR)
    FragBF fa, fb0, fb1;
#pragma unroll
    for (int vv = 0; vv < 8; ++vv) {
      int kb;
      if (vv < 4) kb = half ? (8 + 2 * vv) : (2 * vv);
      else        kb = half ? (24 + 2 * (vv - 4)) : (16 + 2 * (vv - 4));
      v2f pa = *(const v2f*)&Asf[mrow][kb];
      fa.u[vv] = f32_to_bf16_bits(pa.x) | (f32_to_bf16_bits(pa.y) << 16);
      int kc = 2 * vv + (half ? 8 : 0) + (vv >= 4 ? 8 : 0);
      v2f pb0 = *(const v2f*)&Bsf[nc0][kc];
      v2f pb1 = *(const v2f*)&Bsf[nc1][kc];
      fb0.u[vv] = f32_to_bf16_bits(pb0.x) | (f32_to_bf16_bits(pb0.y) << 16);
      fb1.u[vv] = f32_to_bf16_bits(pb1.x) | (f32_to_bf16_bits(pb1.y) << 16);
    }
    acc0 = __builtin_amdgcn_wmma_f32_16x16x32_bf16(false, fa.v, false, fb0.v,
                                                   (short)0, acc0, false, false);
    acc1 = __builtin_amdgcn_wmma_f32_16x16x32_bf16(false, fa.v, false, fb1.v,
                                                   (short)0, acc1, false, false);
    __syncthreads();
  }

  // C/D layout: VGPR r -> M = r (lanes 0-15) or 8+r (lanes 16-31); N = lane%16
  int ncol = lane & 15;
#pragma unroll
  for (int r = 0; r < 8; ++r) {
    int m = m0 + waveM + ((lane < 16) ? r : (r + 8));
    int n = n0 + waveN + ncol;
    if (m < M && n < N) {
      float v = acc0[r] + (bias ? bias[n] : 0.f);
      if (relu) v = fmaxf(v, 0.f);
      C[(size_t)m * N + n] = v;
    }
    int n1 = n + 16;
    if (m < M && n1 < N) {
      float v = acc1[r] + (bias ? bias[n1] : 0.f);
      if (relu) v = fmaxf(v, 0.f);
      C[(size_t)m * N + n1] = v;
    }
  }
}

// ---------------- small utility kernels ----------------
__global__ void fill_f32(float* p, float v, size_t n) {
  size_t i = (size_t)blockIdx.x * blockDim.x + threadIdx.x;
  if (i < n) p[i] = v;
}
__global__ void fill_i32(int* p, int v, size_t n) {
  size_t i = (size_t)blockIdx.x * blockDim.x + threadIdx.x;
  if (i < n) p[i] = v;
}
__global__ void bias_bcast(float* out, const float* b, int C, size_t n) {
  size_t i = (size_t)blockIdx.x * blockDim.x + threadIdx.x;
  if (i < n) out[i] = b[i % C];
}
__global__ void relu_k(float* p, size_t n) {
  size_t i = (size_t)blockIdx.x * blockDim.x + threadIdx.x;
  if (i < n) p[i] = fmaxf(p[i], 0.f);
}
__global__ void rsqrt_k(const float* a, float* o, int n) {
  int i = blockIdx.x * blockDim.x + threadIdx.x;
  if (i < n) o[i] = rsqrtf(a[i]);
}
__global__ void small_linear(const float* X, const float* W, const float* bias,
                             float* Y, int M, int N, int K, int relu) {
  int i = blockIdx.x * blockDim.x + threadIdx.x;
  if (i >= M * N) return;
  int m = i / N, n = i % N;
  float s = bias ? bias[n] : 0.f;
  for (int k2 = 0; k2 < K; ++k2) s += X[m * K + k2] * W[n * K + k2];
  if (relu) s = fmaxf(s, 0.f);
  Y[i] = s;
}

// ---------------- GAT kernels ----------------
__global__ void gat_alpha(const float* __restrict__ h, const float* __restrict__ a_src,
                          const float* __restrict__ a_dst, float* asrc, float* adst, int n) {
  int i = blockIdx.x * blockDim.x + threadIdx.x;  // node*4 + head
  if (i >= n * 4) return;
  int node = i >> 2, hd = i & 3;
  const float* hp = h + (size_t)node * 256 + hd * 64;
  const float* as = a_src + hd * 64;
  const float* ad = a_dst + hd * 64;
  float s = 0.f, d = 0.f;
#pragma unroll 4
  for (int c = 0; c < 64; ++c) { float v = hp[c]; s += v * as[c]; d += v * ad[c]; }
  asrc[i] = s; adst[i] = d;
}

__global__ void gat_edge_max(const int* __restrict__ src, const int* __restrict__ dst,
                             const float* __restrict__ asrc, const float* __restrict__ adst,
                             float* emax, int E, int n) {
  int i = blockIdx.x * blockDim.x + threadIdx.x;
  if (i >= E + n) return;
  int s = (i < E) ? src[i] : (i - E);
  int d = (i < E) ? dst[i] : (i - E);
#pragma unroll
  for (int hd = 0; hd < 4; ++hd) {
    float e = asrc[s * 4 + hd] + adst[d * 4 + hd];
    e = (e > 0.f) ? e : NEG_SLOPE * e;
    atomicMaxFloat(&emax[d * 4 + hd], e);
  }
}

__global__ void gat_edge_exp(const int* __restrict__ src, const int* __restrict__ dst,
                             const float* __restrict__ asrc, const float* __restrict__ adst,
                             const float* __restrict__ emax, float* ee, float* denom,
                             int E, int n) {
  int i = blockIdx.x * blockDim.x + threadIdx.x;
  if (i >= E + n) return;
  int s = (i < E) ? src[i] : (i - E);
  int d = (i < E) ? dst[i] : (i - E);
#pragma unroll
  for (int hd = 0; hd < 4; ++hd) {
    float e = asrc[s * 4 + hd] + adst[d * 4 + hd];
    e = (e > 0.f) ? e : NEG_SLOPE * e;
    float ex = __expf(e - emax[d * 4 + hd]);
    ee[(size_t)i * 4 + hd] = ex;
    atomicAdd(&denom[d * 4 + hd], ex);
  }
}

// one wave32 per edge; 8 features/lane over the 256-wide feature vector
__global__ __launch_bounds__(256) void gat_aggregate(
    const int* __restrict__ src, const int* __restrict__ dst,
    const float* __restrict__ ee, const float* __restrict__ denom,
    const float* __restrict__ h, float* out, int E, int n) {
  int gw = (blockIdx.x * blockDim.x + threadIdx.x) >> 5;
  int lane = threadIdx.x & 31;
  if (gw >= E + n) return;
  int s = (gw < E) ? src[gw] : (gw - E);
  int d = (gw < E) ? dst[gw] : (gw - E);
  float alpha[4];
#pragma unroll
  for (int hd = 0; hd < 4; ++hd)
    alpha[hd] = ee[(size_t)gw * 4 + hd] / (denom[d * 4 + hd] + 1e-16f);
  const float* hs = h + (size_t)s * 256;
  __builtin_prefetch(hs, 0, 0);  // global_prefetch_b8
  float* od = out + (size_t)d * 256;
#pragma unroll
  for (int j = 0; j < 8; ++j) {
    int f = lane + j * 32;
    atomicAdd(&od[f], alpha[f >> 6] * hs[f]);
  }
}

// ---------------- GCN kernels ----------------
__global__ void gcn_deg(const int* __restrict__ dst, const float* __restrict__ mask,
                        float* deg, int E) {
  int i = blockIdx.x * blockDim.x + threadIdx.x;
  if (i < E) atomicAdd(&deg[dst[i]], mask ? mask[i] : 1.f);
}
__global__ void gcn_selfinit(const float* __restrict__ hw, const float* __restrict__ dinv,
                             const float* __restrict__ b, float* out, int n, int C) {
  size_t i = (size_t)blockIdx.x * blockDim.x + threadIdx.x;
  if (i >= (size_t)n * C) return;
  int node = (int)(i / C), c = (int)(i % C);
  float di = dinv[node];
  out[i] = di * di * hw[i] + b[c];
}
__global__ __launch_bounds__(256) void gcn_aggregate(
    const int* __restrict__ src, const int* __restrict__ dst,
    const float* __restrict__ mask, const float* __restrict__ dinv,
    const float* __restrict__ hw, float* out, int E, int C) {
  int gw = (blockIdx.x * blockDim.x + threadIdx.x) >> 5;
  int lane = threadIdx.x & 31;
  if (gw >= E) return;
  int s = src[gw], d = dst[gw];
  float m = mask ? mask[gw] : 1.f;
  float coef = dinv[s] * dinv[d] * m;
  if (coef == 0.f) return;
  const float* hs = hw + (size_t)s * C;
  float* od = out + (size_t)d * C;
  for (int j = 0; j < C / 32; ++j) {
    int f = lane + j * 32;
    atomicAdd(&od[f], coef * hs[f]);
  }
}

// ---------------- SAGPool kernels ----------------
__global__ __launch_bounds__(256) void sag_edge_agg(
    const int* __restrict__ src, const int* __restrict__ dst,
    const float* __restrict__ mask, const float* __restrict__ x,
    float* agg, int E, int C) {
  int gw = (blockIdx.x * blockDim.x + threadIdx.x) >> 5;
  int lane = threadIdx.x & 31;
  if (gw >= E) return;
  float m = mask ? mask[gw] : 1.f;
  if (m == 0.f) return;
  int s = src[gw], d = dst[gw];
  const float* xs = x + (size_t)s * C;
  float* od = agg + (size_t)d * C;
  for (int j = 0; j < C / 32; ++j) {
    int f = lane + j * 32;
    atomicAdd(&od[f], m * xs[f]);
  }
}
__global__ void sag_score(const float* __restrict__ agg, const float* __restrict__ x,
                          const float* __restrict__ wrel, const float* __restrict__ brel,
                          const float* __restrict__ wroot, float* score, int n, int C) {
  int i = blockIdx.x * blockDim.x + threadIdx.x;
  if (i >= n) return;
  float s = brel[0];
  for (int c = 0; c < C; ++c)
    s += agg[(size_t)i * C + c] * wrel[c] + x[(size_t)i * C + c] * wroot[c];
  score[i] = s;
}
// one block per graph: bitonic sort (descending) of npg scores, emit top-k
__global__ __launch_bounds__(1024) void sag_topk(
    const float* __restrict__ score, int* perm, float* scale, int* new_id,
    int npg, int k) {
  __shared__ float ss[2048];
  __shared__ int   si[2048];
  int g = blockIdx.x;
  int tid = threadIdx.x;
  for (int i = tid; i < npg; i += blockDim.x) { ss[i] = score[g * npg + i]; si[i] = i; }
  __syncthreads();
  for (int len = 2; len <= npg; len <<= 1) {
    for (int stride = len >> 1; stride > 0; stride >>= 1) {
      for (int t = tid; t < npg / 2; t += blockDim.x) {
        int i = 2 * t - (t & (stride - 1));
        int j = i + stride;
        bool up = ((i & len) != 0);
        float a = ss[i], b = ss[j];
        if ((a < b) != up) {
          ss[i] = b; ss[j] = a;
          int ti = si[i]; si[i] = si[j]; si[j] = ti;
        }
      }
      __syncthreads();
    }
  }
  for (int t = tid; t < k; t += blockDim.x) {
    int old_node = g * npg + si[t];
    int new_node = g * k + t;
    new_id[old_node] = new_node;
    perm[new_node]   = old_node;
    scale[new_node]  = tanhf(ss[t]);
  }
}
__global__ void sag_gather(const float* __restrict__ x, const int* __restrict__ perm,
                           const float* __restrict__ scale, float* x_new, int n_new, int C) {
  size_t i = (size_t)blockIdx.x * blockDim.x + threadIdx.x;
  if (i >= (size_t)n_new * C) return;
  int nn = (int)(i / C), c = (int)(i % C);
  x_new[i] = x[(size_t)perm[nn] * C + c] * scale[nn];
}
__global__ void sag_filter(const int* __restrict__ s_old, const int* __restrict__ d_old,
                           const float* __restrict__ m_old, const int* __restrict__ new_id,
                           int* s_new, int* d_new, float* m_new, int E) {
  int i = blockIdx.x * blockDim.x + threadIdx.x;
  if (i >= E) return;
  int ns = new_id[s_old[i]], nd = new_id[d_old[i]];
  float m = (m_old ? m_old[i] : 1.f) * ((ns >= 0 && nd >= 0) ? 1.f : 0.f);
  s_new[i] = ns > 0 ? ns : 0;
  d_new[i] = nd > 0 ? nd : 0;
  m_new[i] = m;
}

// ---------------- mean pool ----------------
__global__ void mean_pool(const float* __restrict__ x, float* g, int npg, int C) {
  int b = blockIdx.x, c = threadIdx.x;  // blockDim = C
  float s = 0.f;
  for (int i = 0; i < npg; ++i) s += x[((size_t)(b * npg + i)) * C + c];
  g[b * C + c] = s / (float)npg;
}

// =====================================================================
extern "C" void kernel_launch(void* const* d_in, const int* in_sizes, int n_in,
                              void* d_out, int out_size, void* d_ws, size_t ws_size,
                              hipStream_t stream) {
  (void)in_sizes; (void)n_in; (void)out_size; (void)ws_size;
  const float* x          = (const float*)d_in[0];
  const int*   edge_index = (const int*)d_in[1];
  const int*   src = edge_index;
  const int*   dst = edge_index + E_TOT;
  const float* gat1_w = (const float*)d_in[3];
  const float* gat1_as = (const float*)d_in[4];
  const float* gat1_ad = (const float*)d_in[5];
  const float* gat1_b = (const float*)d_in[6];
  const float* gat2_w = (const float*)d_in[7];
  const float* gat2_as = (const float*)d_in[8];
  const float* gat2_ad = (const float*)d_in[9];
  const float* gat2_b = (const float*)d_in[10];
  const float* gcn1_w = (const float*)d_in[11];
  const float* gcn1_b = (const float*)d_in[12];
  const float* sag1_wrel = (const float*)d_in[13];
  const float* sag1_brel = (const float*)d_in[14];
  const float* sag1_wroot = (const float*)d_in[15];
  const float* gcn2_w = (const float*)d_in[16];
  const float* gcn2_b = (const float*)d_in[17];
  const float* sag2_wrel = (const float*)d_in[18];
  const float* sag2_brel = (const float*)d_in[19];
  const float* sag2_wroot = (const float*)d_in[20];
  const float* fc1_w = (const float*)d_in[21];
  const float* fc1_b = (const float*)d_in[22];
  const float* fc2_w = (const float*)d_in[23];
  const float* fc2_b = (const float*)d_in[24];
  const float* out_w = (const float*)d_in[25];
  const float* out_b = (const float*)d_in[26];
  float* out = (float*)d_out;

  // ---- workspace arena ----
  uint8_t* wsp = (uint8_t*)d_ws;
  size_t off = 0;
  auto alloc = [&](size_t nbytes) -> void* {
    void* p = wsp + off;
    off += (nbytes + 255) & ~(size_t)255;
    return p;
  };
  float* hA    = (float*)alloc((size_t)N_TOT * 256 * 4);   // 32 MB (reused heavily)
  float* hB    = (float*)alloc((size_t)N_TOT * 256 * 4);   // 32 MB (reused heavily)
  float* asrc  = (float*)alloc((size_t)N_TOT * 4 * 4);
  float* adst  = (float*)alloc((size_t)N_TOT * 4 * 4);
  float* emax  = (float*)alloc((size_t)N_TOT * 4 * 4);
  float* denom = (float*)alloc((size_t)N_TOT * 4 * 4);
  float* ee    = (float*)alloc((size_t)(E_TOT + N_TOT) * 4 * 4);
  float* deg   = (float*)alloc((size_t)N_TOT * 4);
  float* dinv  = (float*)alloc((size_t)N_TOT * 4);
  int*   newid = (int*)  alloc((size_t)N_TOT * 4);
  float* score = (float*)alloc((size_t)N_TOT * 4);
  float* scl   = (float*)alloc((size_t)N_TOT * 4);
  int*   perm  = (int*)  alloc((size_t)N_TOT * 4);
  int*   s1    = (int*)  alloc((size_t)E_TOT * 4);
  int*   d1    = (int*)  alloc((size_t)E_TOT * 4);
  float* m1    = (float*)alloc((size_t)E_TOT * 4);
  int*   s2    = (int*)  alloc((size_t)E_TOT * 4);
  int*   d2    = (int*)  alloc((size_t)E_TOT * 4);
  float* m2    = (float*)alloc((size_t)E_TOT * 4);
  float* g0    = (float*)alloc(16 * 128 * 4);
  float* g1    = (float*)alloc(16 * 256 * 4);
  float* g2    = (float*)alloc(16 * 128 * 4);

  auto cdiv = [](long long a, long long b) -> int { return (int)((a + b - 1) / b); };

  // ---- GAT layer helper: h = xin @ w^T ; softmax attention ; aggregate ----
  auto run_gat = [&](const float* xin, int Cin, const float* wgt, const float* a_s,
                     const float* a_d, const float* bvec, float* hbuf, float* outbuf,
                     int relu) {
    gemm_bf16_wmma<<<dim3(N_TOT / TM, 256 / TN), 256, 0, stream>>>(
        xin, wgt, nullptr, hbuf, N_TOT, 256, Cin, 0);
    gat_alpha<<<cdiv(N_TOT * 4, 256), 256, 0, stream>>>(hbuf, a_s, a_d, asrc, adst, N_TOT);
    fill_f32<<<cdiv(N_TOT * 4, 256), 256, 0, stream>>>(emax, -1e30f, (size_t)N_TOT * 4);
    fill_f32<<<cdiv(N_TOT * 4, 256), 256, 0, stream>>>(denom, 0.f, (size_t)N_TOT * 4);
    gat_edge_max<<<cdiv(E_TOT + N_TOT, 256), 256, 0, stream>>>(
        src, dst, asrc, adst, emax, E_TOT, N_TOT);
    gat_edge_exp<<<cdiv(E_TOT + N_TOT, 256), 256, 0, stream>>>(
        src, dst, asrc, adst, emax, ee, denom, E_TOT, N_TOT);
    bias_bcast<<<cdiv((long long)N_TOT * 256, 256), 256, 0, stream>>>(
        outbuf, bvec, 256, (size_t)N_TOT * 256);
    gat_aggregate<<<cdiv((long long)(E_TOT + N_TOT) * 32, 256), 256, 0, stream>>>(
        src, dst, ee, denom, hbuf, outbuf, E_TOT, N_TOT);
    if (relu)
      relu_k<<<cdiv((long long)N_TOT * 256, 256), 256, 0, stream>>>(
          outbuf, (size_t)N_TOT * 256);
  };

  // ---- GCN layer helper ----
  auto run_gcn = [&](const float* xin, int n, int Cin, int Cout, const int* s_,
                     const int* d_, const float* m_, const float* wgt,
                     const float* bvec, float* hw, float* outp) {
    gemm_bf16_wmma<<<dim3(cdiv(n, TM), cdiv(Cout, TN)), 256, 0, stream>>>(
        xin, wgt, nullptr, hw, n, Cout, Cin, 0);
    fill_f32<<<cdiv(n, 256), 256, 0, stream>>>(deg, 1.f, (size_t)n);
    gcn_deg<<<cdiv(E_TOT, 256), 256, 0, stream>>>(d_, m_, deg, E_TOT);
    rsqrt_k<<<cdiv(n, 256), 256, 0, stream>>>(deg, dinv, n);
    gcn_selfinit<<<cdiv((long long)n * Cout, 256), 256, 0, stream>>>(
        hw, dinv, bvec, outp, n, Cout);
    gcn_aggregate<<<cdiv((long long)E_TOT * 32, 256), 256, 0, stream>>>(
        s_, d_, m_, dinv, hw, outp, E_TOT, Cout);
    relu_k<<<cdiv((long long)n * Cout, 256), 256, 0, stream>>>(outp, (size_t)n * Cout);
  };

  // ---- SAGPool helper ----
  auto run_sag = [&](const float* xin, int n, int npg, int k, const int* s_,
                     const int* d_, const float* m_, const float* wrel,
                     const float* brel, const float* wroot, float* aggb, float* xnew,
                     int* sn, int* dn, float* mn) {
    fill_f32<<<cdiv((long long)n * 128, 256), 256, 0, stream>>>(aggb, 0.f, (size_t)n * 128);
    sag_edge_agg<<<cdiv((long long)E_TOT * 32, 256), 256, 0, stream>>>(
        s_, d_, m_, xin, aggb, E_TOT, 128);
    sag_score<<<cdiv(n, 256), 256, 0, stream>>>(aggb, xin, wrel, brel, wroot, score, n, 128);
    fill_i32<<<cdiv(n, 256), 256, 0, stream>>>(newid, -1, (size_t)n);
    sag_topk<<<n / npg, 1024, 0, stream>>>(score, perm, scl, newid, npg, k);
    int n_new = (n / npg) * k;
    sag_gather<<<cdiv((long long)n_new * 128, 256), 256, 0, stream>>>(
        xin, perm, scl, xnew, n_new, 128);
    sag_filter<<<cdiv(E_TOT, 256), 256, 0, stream>>>(s_, d_, m_, newid, sn, dn, mn, E_TOT);
  };

  // ============ pipeline ============
  // GAT1: x[32768,64] -> hB[32768,256], relu
  run_gat(x, 64, gat1_w, gat1_as, gat1_ad, gat1_b, hA, hB, 1);
  // GAT2: hB -> hA (h2), aggregate back into hB (gat2 input dead after gemm)
  run_gat(hB, 256, gat2_w, gat2_as, gat2_ad, gat2_b, hA, hB, 0);

  // GCN1: hB[32768,256] -> hA regions
  float* gcn1h   = hA;                              // [32768,128]
  float* gcn1out = hA + (size_t)N_TOT * 128;        // [32768,128]
  run_gcn(hB, N_TOT, 256, 128, src, dst, nullptr, gcn1_w, gcn1_b, gcn1h, gcn1out);

  // SAGPool1: 32768 -> 16384 nodes (k=1024/graph)
  float* x1 = hB;                                   // [16384,128]
  run_sag(gcn1out, N_TOT, NPG, NPG / 2, src, dst, nullptr,
          sag1_wrel, sag1_brel, sag1_wroot, /*agg*/ gcn1h, x1, s1, d1, m1);

  // GCN2: x1[16384,128]
  int n2 = B_GRAPHS * (NPG / 2);                    // 16384
  float* gcn2h   = hA;                              // [16384,128]
  float* gcn2out = hA + (size_t)n2 * 128;           // [16384,128]
  run_gcn(x1, n2, 128, 128, s1, d1, m1, gcn2_w, gcn2_b, gcn2h, gcn2out);

  // SAGPool2: 16384 -> 8192 nodes (k=512/graph)
  float* agg2 = hA + (size_t)2 * n2 * 128;
  float* x2   = hB + (size_t)n2 * 128;              // [8192,128]
  run_sag(gcn2out, n2, NPG / 2, NPG / 4, s1, d1, m1,
          sag2_wrel, sag2_brel, sag2_wroot, agg2, x2, s2, d2, m2);

  // global mean pool: [16,512,128] -> g0[16,128]
  mean_pool<<<B_GRAPHS, 128, 0, stream>>>(x2, g0, NPG / 4, 128);

  // classifier MLP (WMMA for fc1/fc2, tiny VALU kernel for 10-wide output)
  gemm_bf16_wmma<<<dim3(1, 4), 256, 0, stream>>>(g0, fc1_w, fc1_b, g1, 16, 256, 128, 1);
  gemm_bf16_wmma<<<dim3(1, 2), 256, 0, stream>>>(g1, fc2_w, fc2_b, g2, 16, 128, 256, 1);
  small_linear<<<1, 256, 0, stream>>>(g2, out_w, out_b, out, 16, 10, 128, 0);
}
